// VectorQuantize_24189255811230
// MI455X (gfx1250) — matrix operand order repaired and checked by reading, not compiled, not run on hardware
//
#include <hip/hip_runtime.h>

// Problem constants (match reference)
#define B_        4
#define N_        2048
#define DIM_      256
#define HEADS_    4
#define CODEBOOK_ 8192
#define HD_       64
#define BN_       (B_ * N_)          // 8192 tokens
#define QSIZE_    (B_ * N_ * DIM_)   // 2,097,152 floats (quantize output)

typedef __attribute__((ext_vector_type(16))) __bf16 v16bf;
typedef __attribute__((ext_vector_type(8)))  float  v8f;

union V16U { v16bf v; uint4  q[2]; };   // 32 bytes
union V8FU { v8f   v; float4 f[2]; };   // 32 bytes

// ---------------------------------------------------------------------------
// gfx1250 async copy: global -> LDS, 16 bytes/lane, GVS addressing
// (SGPR 64-bit base + per-lane 32-bit byte offset). Tracked by ASYNCcnt.
// ---------------------------------------------------------------------------
__device__ __forceinline__ void async_copy_b128(unsigned lds_byte_addr,
                                                unsigned global_byte_off,
                                                unsigned long long sbase) {
  asm volatile("global_load_async_to_lds_b128 %0, %1, %2"
               :: "v"(lds_byte_addr), "v"(global_byte_off), "s"(sbase)
               : "memory");
}
__device__ __forceinline__ void wait_async0() {
  asm volatile("s_wait_asynccnt 0x0" ::: "memory");
}
__device__ __forceinline__ unsigned lds_addr_of(const void* p) {
  return (unsigned)(unsigned long long)p;   // low 32 bits = LDS byte offset
}

// ---------------------------------------------------------------------------
// Kernel 1 (one-time prep): split embed into planar bf16 hi/lo and compute
// pre-scaled bias  y2n[h,c] = -0.5 * sum_k embed[h,c,k]^2.
// ---------------------------------------------------------------------------
__global__ __launch_bounds__(256)
void vq_prep_kernel(const float* __restrict__ embed,
                    __bf16* __restrict__ ehi, __bf16* __restrict__ elo,
                    float* __restrict__ y2n) {
  const int c = blockIdx.x * 256 + threadIdx.x;       // 0 .. HEADS*CODEBOOK-1
  const float4* row = (const float4*)(embed + (size_t)c * HD_);
  __bf16* hrow = ehi + (size_t)c * HD_;
  __bf16* lrow = elo + (size_t)c * HD_;
  float s = 0.f;
#pragma unroll
  for (int i = 0; i < HD_ / 4; ++i) {
    const float4 e = row[i];
    const float vv[4] = {e.x, e.y, e.z, e.w};
#pragma unroll
    for (int j = 0; j < 4; ++j) {
      const float  v  = vv[j];
      const __bf16 hb = (__bf16)v;
      hrow[i * 4 + j] = hb;
      lrow[i * 4 + j] = (__bf16)(v - (float)hb);
      s += v * v;
    }
  }
  y2n[c] = -0.5f * s;
}

// ---------------------------------------------------------------------------
// Kernel 2: fused GEMM + argmax + gather.
//   grid = (BN/128 token tiles, HEADS) ; block = 256 (8 waves)
//   wave w owns tokens [tokbase + 16w, +16); sweeps all 8192 codes in
//   double-buffered 64-code stages brought in with async-LDS copies.
//   WMMA tile: D[M=code(16), N=token(16)] = A(codes 16xK) x B(tokens Kx16) + C
//   C seeded with precomputed -0.5*||y||^2 so D is the score x.y - 0.5*y2.
// ---------------------------------------------------------------------------
__global__ __launch_bounds__(256)
void vq_argmax_kernel(const float* __restrict__ x,
                      const float* __restrict__ embed,
                      const __bf16* __restrict__ ehi,
                      const __bf16* __restrict__ elo,
                      const float* __restrict__ y2n,
                      float* __restrict__ out) {
  const int h       = blockIdx.y;
  const int tokbase = blockIdx.x * 128;
  const int tid     = threadIdx.x;
  const int wave    = tid >> 5;
  const int lane    = tid & 31;
  const int col     = lane & 15;          // token column / code row in tile
  const int half8   = (lane >> 4) * 8;    // 0 or 8 (K-segment / M-row offset)

  // Double-buffered stage: 64 codes x 64 bf16 per plane (8 KB/plane/buf)
  __shared__ __align__(16) __bf16 s_hi[2][64 * 64];
  __shared__ __align__(16) __bf16 s_lo[2][64 * 64];
  __shared__ __align__(16) float  s_y2[2][64];

  // ---- Load this wave's 16 tokens as B operands (resident all sweep) ----
  // B 32x16 bf16 in this code's k order (same as A): lane holds column N=lane%16; lanes 0-15 K=0..15,
  // lanes 16-31 K=16..31 (consecutive within the v16bf).
  const int    mtok = tokbase + wave * 16 + col;
  const float* xrow = x + (size_t)mtok * DIM_ + h * HD_;
  const int    koff = (lane < 16) ? 0 : 16;

  V16U bhi0, blo0, bhi1, blo1;
  {
    auto loadB = [&](int kbase, V16U& hi, V16U& lo) {
      const float4* p = (const float4*)(xrow + kbase);
#pragma unroll
      for (int q = 0; q < 4; ++q) {
        const float4 e = p[q];
        const float vv[4] = {e.x, e.y, e.z, e.w};
#pragma unroll
        for (int i = 0; i < 4; ++i) {
          const __bf16 hb = (__bf16)vv[i];
          hi.v[q * 4 + i] = hb;
          lo.v[q * 4 + i] = (__bf16)(vv[i] - (float)hb);
        }
      }
    };
    loadB(koff,      bhi0, blo0);   // K chunk 0..31
    loadB(32 + koff, bhi1, blo1);   // K chunk 32..63
  }

  const unsigned long long ehi_b = (unsigned long long)ehi;
  const unsigned long long elo_b = (unsigned long long)elo;
  const unsigned long long y2_b  = (unsigned long long)y2n;
  const unsigned head_row0 = (unsigned)(h * CODEBOOK_);

  // Issue async stage t into buffer t&1: 8KB/plane = 2 x b128 per thread.
  auto issue_stage = [&](int t) {
    const int      buf = t & 1;
    const unsigned gb  = (head_row0 + (unsigned)(t * 64)) * (HD_ * 2); // bytes
    const unsigned lhi = lds_addr_of(&s_hi[buf][0]) + (unsigned)tid * 16u;
    const unsigned llo = lds_addr_of(&s_lo[buf][0]) + (unsigned)tid * 16u;
    const unsigned go  = gb + (unsigned)tid * 16u;
    async_copy_b128(lhi,          go,          ehi_b);
    async_copy_b128(lhi + 4096u,  go + 4096u,  ehi_b);
    async_copy_b128(llo,          go,          elo_b);
    async_copy_b128(llo + 4096u,  go + 4096u,  elo_b);
    if (tid < 16) {  // 64 bias floats = 256 B
      const unsigned ly = lds_addr_of(&s_y2[buf][0]) + (unsigned)tid * 16u;
      async_copy_b128(ly, (head_row0 + (unsigned)(t * 64)) * 4u + (unsigned)tid * 16u, y2_b);
    }
  };

  float best = -3.4e38f;
  int   bidx = 0;

  const int NSTAGE = CODEBOOK_ / 64;   // 128
  issue_stage(0);

  for (int t = 0; t < NSTAGE; ++t) {
    wait_async0();        // my async writes for buf t&1 are done
    __syncthreads();      // everyone's are; and buf (t+1)&1 is free again
    if (t + 1 < NSTAGE) issue_stage(t + 1);

    const int bufc = t & 1;
    const int cb0  = t * 64;

#pragma unroll
    for (int sub = 0; sub < 4; ++sub) {
      // A operands from LDS (16-bit A layout: two contiguous 8-halfword runs)
      // lane row M = lane%16; elems 0..15 = K 2*half8.. (contiguous, matches B)
      const __bf16* hp = &s_hi[bufc][(sub * 16 + col) * 64];
      const __bf16* lp = &s_lo[bufc][(sub * 16 + col) * 64];
      V16U ah0, al0, ah1, al1;
      ah0.q[0] = *(const uint4*)(hp +      2 * half8);
      ah0.q[1] = *(const uint4*)(hp + 8 +  2 * half8);
      ah1.q[0] = *(const uint4*)(hp + 32 + 2 * half8);
      ah1.q[1] = *(const uint4*)(hp + 40 + 2 * half8);
      al0.q[0] = *(const uint4*)(lp +      2 * half8);
      al0.q[1] = *(const uint4*)(lp + 8 +  2 * half8);
      al1.q[0] = *(const uint4*)(lp + 32 + 2 * half8);
      al1.q[1] = *(const uint4*)(lp + 40 + 2 * half8);

      // C init: bias already equals -0.5*y2 (VGPR r -> code row half8+r)
      V8FU acc;
      acc.f[0] = *(const float4*)(&s_y2[bufc][sub * 16 + half8]);
      acc.f[1] = *(const float4*)(&s_y2[bufc][sub * 16 + half8 + 4]);

      // 6x v_wmma_f32_16x16x32_bf16: (ah+al).(bh+bl) ~ ahbh + ahbl + albh
      acc.v = __builtin_amdgcn_wmma_f32_16x16x32_bf16(false, ah0.v, false, bhi0.v, (short)0, acc.v, false, false);
      acc.v = __builtin_amdgcn_wmma_f32_16x16x32_bf16(false, ah1.v, false, bhi1.v, (short)0, acc.v, false, false);
      acc.v = __builtin_amdgcn_wmma_f32_16x16x32_bf16(false, ah0.v, false, blo0.v, (short)0, acc.v, false, false);
      acc.v = __builtin_amdgcn_wmma_f32_16x16x32_bf16(false, ah1.v, false, blo1.v, (short)0, acc.v, false, false);
      acc.v = __builtin_amdgcn_wmma_f32_16x16x32_bf16(false, al0.v, false, bhi0.v, (short)0, acc.v, false, false);
      acc.v = __builtin_amdgcn_wmma_f32_16x16x32_bf16(false, al1.v, false, bhi1.v, (short)0, acc.v, false, false);

      // lane-local running argmax over its 8 codes of this sub-tile
#pragma unroll
      for (int r = 0; r < 8; ++r) {
        const float s   = acc.v[r];
        const int   idx = cb0 + sub * 16 + half8 + r;
        if (s > best) { best = s; bidx = idx; }   // strict > keeps first max
      }
    }
  }

  // ---- merge the two lane halves tracking the same token ----
  {
    const float os = __shfl_xor(best, 16, 32);
    const int   oi = __shfl_xor(bidx, 16, 32);
    if (os > best || (os == best && oi < bidx)) { best = os; bidx = oi; }
  }

  // ---- gather winning codes (fp32) + write indices ----
  const float* ebase = embed + (size_t)h * CODEBOOK_ * HD_;
#pragma unroll 1
  for (int j = 0; j < 16; ++j) {
    const int cj = __shfl(bidx, j, 32);
    const int m  = tokbase + wave * 16 + j;
    const float2 v = *(const float2*)(ebase + (size_t)cj * HD_ + lane * 2);
    *(float2*)(out + (size_t)m * DIM_ + h * HD_ + lane * 2) = v;
  }
  if (lane < 16) {
    const int m = tokbase + wave * 16 + lane;
    out[QSIZE_ + (size_t)m * HEADS_ + h] = (float)bidx;
  }
}

// ---------------------------------------------------------------------------
extern "C" void kernel_launch(void* const* d_in, const int* in_sizes, int n_in,
                              void* d_out, int out_size, void* d_ws, size_t ws_size,
                              hipStream_t stream) {
  const float* x     = (const float*)d_in[0];   // [B, N, DIM] fp32
  const float* embed = (const float*)d_in[1];   // [HEADS, CODEBOOK, HD] fp32

  // Workspace layout: ehi (4 MB bf16) ++ elo (4 MB bf16) ++ y2n (128 KB f32)
  __bf16* ehi = (__bf16*)d_ws;
  __bf16* elo = ehi + (size_t)HEADS_ * CODEBOOK_ * HD_;
  float*  y2n = (float*)(elo + (size_t)HEADS_ * CODEBOOK_ * HD_);
  float*  out = (float*)d_out;

  vq_prep_kernel<<<dim3((HEADS_ * CODEBOOK_) / 256), 256, 0, stream>>>(embed, ehi, elo, y2n);
  vq_argmax_kernel<<<dim3(BN_ / 128, HEADS_), 256, 0, stream>>>(x, embed, ehi, elo, y2n, out);
}
